// PIMGT_External_14903536517748
// MI455X (gfx1250) — matrix operand order repaired and checked by reading, not compiled, not run on hardware
//
#include <hip/hip_runtime.h>
#include <hip/hip_bf16.h>

#define DEVFN __device__ __forceinline__

#if __has_include(<hip/amd_detail/amd_gfx1250_TDM.h>)
#define TDM_6ARG 1
#else
#define TDM_6ARG 0
#endif

#if defined(__AMDGCN__) && __has_builtin(__builtin_amdgcn_tensor_load_to_lds)
#define HAVE_TDM 1
#else
#define HAVE_TDM 0
#endif

typedef __bf16 bf16;
typedef __attribute__((ext_vector_type(8)))  bf16  v8bf;
typedef __attribute__((ext_vector_type(16))) bf16  v16bf;
typedef __attribute__((ext_vector_type(8)))  float v8f;
typedef __attribute__((ext_vector_type(2)))  float v2f;
typedef __attribute__((ext_vector_type(4)))  unsigned int v4u;
typedef __attribute__((ext_vector_type(8)))  int   v8i;
typedef __attribute__((ext_vector_type(4)))  int   v4i;

static constexpr int  B_ = 32, P_ = 12, N_ = 512, F_ = 64, D_ = 256, Q_ = 12;
static constexpr float DT_ = 0.1f;
static constexpr long long MROWS = (long long)B_ * Q_ * N_;   // 196608

// macro-tile geometry for the LDS-tiled GEMM
static constexpr int BM = 128, BN = 128, BK = 32;

// ---------------------------------------------------------------------------
// bf16 fragment loaders (ISA 16-bit A-layout: lanes 0-15 K=0..7,16..23;
// lanes 16-31 K=8..15,24..31 — two 16B vector loads per fragment)
// ---------------------------------------------------------------------------
DEVFN v16bf frag2(const bf16* p0, const bf16* p1) {
    union { v16bf v; v8bf h[2]; } u;
    u.h[0] = *reinterpret_cast<const v8bf*>(p0);
    u.h[1] = *reinterpret_cast<const v8bf*>(p1);
    return u.v;
}

DEVFN v16bf load_a_bf16(const bf16* A, long long row, int lda, int k0, int lane) {
    int r = lane & 15, hh = lane >> 4;
    const bf16* p0 = A + (row + r) * (long long)lda + k0 + hh * 8;
    return frag2(p0, p0 + 16);
}

DEVFN v16bf load_b_bf16(const bf16* Bt, int tn, int ldb, int k0, int lane) {
    int c = lane & 15, hh = lane >> 4;
    const bf16* p0 = Bt + (long long)(tn + c) * ldb + k0 + hh * 8;
    return frag2(p0, p0 + 16);
}

// LDS variants (tile row stride = BK elements)
DEVFN v16bf lds_a_frag(const bf16* tile, int row0, int lane) {
    int r = lane & 15, hh = lane >> 4;
    const bf16* p0 = tile + (row0 + r) * BK + hh * 8;
    return frag2(p0, p0 + 16);
}
DEVFN v16bf lds_b_frag(const bf16* tile, int col0, int lane) {
    int c = lane & 15, hh = lane >> 4;
    const bf16* p0 = tile + (col0 + c) * BK + hh * 8;
    return frag2(p0, p0 + 16);
}

// ---------------------------------------------------------------------------
// Tensor Data Mover: async-load a rows x rowlen bf16 tile (row stride =
// stride_elems) from global into LDS at lds_byte_off.  D# packed per CDNA5
// ISA 08_async_tensor §8 (group0: count/lds/global/type; group1: data_size,
// tile dims, tensor_dim0_stride).  Issued wave-uniformly; tracked by TENSORcnt.
// ---------------------------------------------------------------------------
#if HAVE_TDM
DEVFN void tdm_load_tile_bf16(unsigned lds_byte_off, const bf16* gptr,
                              int rows, int rowlen, long long stride_elems)
{
    unsigned long long ga = (unsigned long long)gptr;
    v4u g0;
    g0[0] = 1u;                                               // count=1
    g0[1] = lds_byte_off;                                     // lds_addr
    g0[2] = (unsigned)(ga & 0xffffffffu);                     // global_addr lo
    g0[3] = (unsigned)((ga >> 32) & 0x01ffffffu) | (2u << 30);// addr hi | type=2
    v8i g1;
    unsigned tdim0 = 0x7fffffffu, tdim1 = 0x7fffffffu;        // no OOB clip
    unsigned long long s0 = (unsigned long long)stride_elems; // dim0 stride
    g1[0] = (int)(1u << 16);                                  // data_size=2B
    g1[1] = (int)((tdim0 & 0xffffu) << 16);                   // tensor_dim0 lo
    g1[2] = (int)((tdim0 >> 16) & 0xffffu) |
            (int)((tdim1 & 0xffffu) << 16);                   // dim0 hi | dim1 lo
    g1[3] = (int)((tdim1 >> 16) & 0xffffu) |
            (int)((unsigned)rowlen << 16);                    // dim1 hi | tile_dim0
    g1[4] = (int)(unsigned)rows;                              // tile_dim1
    g1[5] = (int)(unsigned)(s0 & 0xffffffffu);                // dim0_stride lo
    g1[6] = (int)(unsigned)((s0 >> 32) & 0xffffu);            // dim0_stride hi
    g1[7] = 0;
    v4i z4 = {0, 0, 0, 0};
#if TDM_6ARG
    v8i z8 = {0, 0, 0, 0, 0, 0, 0, 0};
    __builtin_amdgcn_tensor_load_to_lds(g0, g1, z4, z4, z8, 0);
#else
    __builtin_amdgcn_tensor_load_to_lds(g0, g1, z4, z4, 0);
#endif
}
DEVFN void wait_tensorcnt_2() { __builtin_amdgcn_s_wait_tensorcnt(2); }
DEVFN void wait_tensorcnt_0() { __builtin_amdgcn_s_wait_tensorcnt(0); }
#endif

// cooperative fallback tile fill (256 threads, 128x32 bf16)
DEVFN void fill_tile(bf16* dst, const bf16* src, long long stride) {
    for (int c = threadIdx.x; c < (BM * BK) / 8; c += 256) {
        int row = c >> 2, chunk = c & 3;
        *reinterpret_cast<v8bf*>(dst + row * BK + chunk * 8) =
            *reinterpret_cast<const v8bf*>(src + (long long)row * stride + chunk * 8);
    }
}

// ---------------------------------------------------------------------------
// LDS-tiled batched bf16 WMMA GEMM: C[z] = A[z] (MxK) * Bt[z]^T (Nc x K).
// Macro tile 128x128, 8 waves x (2x4) 16x16 accumulators, BK=32 K-steps,
// double-buffered LDS fed by TENSOR_LOAD_TO_LDS.  M,Nc mult of 128; K of 32.
// ---------------------------------------------------------------------------
__global__ __launch_bounds__(256) void k_gemm_bf16_tiled(
    const bf16* __restrict__ A,  int lda, long long strideA,
    const bf16* __restrict__ Bt, int ldb, long long strideB,
    float* __restrict__ Cf,  long long strideCf,
    bf16*  __restrict__ Cb,  long long strideCb,
    bf16*  __restrict__ CbT, int segRows,
    const float* __restrict__ addF, long long strideAdd,
    long long M, int Nc, int K)
{
    __shared__ bf16 As[2][BM * BK];
    __shared__ bf16 Bs[2][BN * BK];

    const int wave = threadIdx.x >> 5, lane = threadIdx.x & 31;
    const int wm = wave >> 1;            // 0..3  -> 32-row strip
    const int wn = wave & 1;             // 0..1  -> 64-col strip

    const int tilesN = Nc / BN;
    const long long bm = blockIdx.x / tilesN;
    const int       bn = (int)(blockIdx.x % tilesN);
    const long long rowBase = bm * BM;
    const int       colBase = bn * BN;

    const bf16* Az = A  + (long long)blockIdx.z * strideA + rowBase * lda;
    const bf16* Bz = Bt + (long long)blockIdx.z * strideB + (long long)colBase * ldb;

    v8f acc[2][4];
    for (int i = 0; i < 2; ++i)
        for (int j = 0; j < 4; ++j)
            acc[i][j] = v8f{0.f, 0.f, 0.f, 0.f, 0.f, 0.f, 0.f, 0.f};

    const int KT = K / BK;

#if HAVE_TDM
    const unsigned aOff[2] = { (unsigned)(unsigned long long)&As[0][0],
                               (unsigned)(unsigned long long)&As[1][0] };
    const unsigned bOff[2] = { (unsigned)(unsigned long long)&Bs[0][0],
                               (unsigned)(unsigned long long)&Bs[1][0] };
    if (threadIdx.x < 32) {
        tdm_load_tile_bf16(aOff[0], Az, BM, BK, lda);
        tdm_load_tile_bf16(bOff[0], Bz, BN, BK, ldb);
    }
    for (int kt = 0; kt < KT; ++kt) {
        if (threadIdx.x < 32) {
            if (kt + 1 < KT) {
                tdm_load_tile_bf16(aOff[(kt + 1) & 1], Az + (long long)(kt + 1) * BK,
                                   BM, BK, lda);
                tdm_load_tile_bf16(bOff[(kt + 1) & 1], Bz + (long long)(kt + 1) * BK,
                                   BN, BK, ldb);
                wait_tensorcnt_2();       // current buffer's pair has landed
            } else {
                wait_tensorcnt_0();
            }
        }
        __syncthreads();
        const bf16* as = As[kt & 1];
        const bf16* bs = Bs[kt & 1];
        v16bf af[2], bfrag[4];
        for (int i = 0; i < 2; ++i) af[i] = lds_a_frag(as, wm * 32 + i * 16, lane);
        for (int j = 0; j < 4; ++j) bfrag[j] = lds_b_frag(bs, wn * 64 + j * 16, lane);
        for (int i = 0; i < 2; ++i)
            for (int j = 0; j < 4; ++j)
                acc[i][j] = __builtin_amdgcn_wmma_f32_16x16x32_bf16(
                    false, af[i], false, bfrag[j], (short)0, acc[i][j], false, false);
        __syncthreads();
    }
#else
    for (int kt = 0; kt < KT; ++kt) {
        __syncthreads();
        fill_tile(As[0], Az + (long long)kt * BK, lda);
        fill_tile(Bs[0], Bz + (long long)kt * BK, ldb);
        __syncthreads();
        v16bf af[2], bfrag[4];
        for (int i = 0; i < 2; ++i) af[i] = lds_a_frag(As[0], wm * 32 + i * 16, lane);
        for (int j = 0; j < 4; ++j) bfrag[j] = lds_b_frag(Bs[0], wn * 64 + j * 16, lane);
        for (int i = 0; i < 2; ++i)
            for (int j = 0; j < 4; ++j)
                acc[i][j] = __builtin_amdgcn_wmma_f32_16x16x32_bf16(
                    false, af[i], false, bfrag[j], (short)0, acc[i][j], false, false);
    }
#endif

    // epilogue
    const int r = lane & 15, hh = lane >> 4;
    for (int i = 0; i < 2; ++i)
        for (int j = 0; j < 4; ++j)
            for (int e = 0; e < 8; ++e) {
                long long row = rowBase + wm * 32 + i * 16 + hh * 8 + e;
                int       col = colBase + wn * 64 + j * 16 + r;
                float val = acc[i][j][e];
                if (addF) val += addF[(long long)blockIdx.z * strideAdd + row * Nc + col];
                if (Cf)   Cf[(long long)blockIdx.z * strideCf + row * Nc + col] = val;
                if (Cb)   Cb[(long long)blockIdx.z * strideCb + row * Nc + col] = (bf16)val;
                if (CbT) {
                    long long s = row / segRows, n = row % segRows;
                    CbT[(s * Nc + col) * segRows + n] = (bf16)val;
                }
            }
}

// ---------------------------------------------------------------------------
// Simple per-wave bf16 GEMM (kept for Nc=64 output projection)
// ---------------------------------------------------------------------------
__global__ __launch_bounds__(256) void k_gemm_bf16(
    const bf16* __restrict__ A,  int lda,
    const bf16* __restrict__ Bt, int ldb,
    float* __restrict__ Cf,
    long long M, int Nc, int K)
{
    int wave = threadIdx.x >> 5, lane = threadIdx.x & 31;
    long long tile   = (long long)blockIdx.x * 8 + wave;
    long long tilesN = Nc / 16;
    if (tile >= (M / 16) * tilesN) return;
    int       tn = (int)(tile % tilesN) * 16;
    long long tm = (tile / tilesN) * 16;

    v8f acc = {0.f, 0.f, 0.f, 0.f, 0.f, 0.f, 0.f, 0.f};
    for (int k0 = 0; k0 < K; k0 += 32) {
        if (k0 + 64 < K)
            __builtin_prefetch(A + (tm + (lane & 15)) * (long long)lda + k0 + 64, 0, 3);
        v16bf a = load_a_bf16(A, tm, lda, k0, lane);
        v16bf b = load_b_bf16(Bt, tn, ldb, k0, lane);
        acc = __builtin_amdgcn_wmma_f32_16x16x32_bf16(
            false, a, false, b, (short)0, acc, false, false);
    }
    int r = lane & 15, hh = lane >> 4;
    for (int j = 0; j < 8; ++j) {
        long long row = tm + hh * 8 + j;
        Cf[row * Nc + tn + r] = acc[j];
    }
}

// ---------------------------------------------------------------------------
// f32 WMMA GEMM (16x16x4): Kalman propagator chain (precision-critical)
// ---------------------------------------------------------------------------
__global__ __launch_bounds__(256) void k_gemm_f32(
    const float* __restrict__ A, int lda, long long strideA,
    const float* __restrict__ B, int ldb, long long strideB,
    float* __restrict__ C, int ldc, long long strideC,
    long long M, int Nc, int K)
{
    int wave = threadIdx.x >> 5, lane = threadIdx.x & 31;
    long long tile   = (long long)blockIdx.x * 8 + wave;
    long long tilesN = Nc / 16;
    if (tile >= (M / 16) * tilesN) return;
    int       tn = (int)(tile % tilesN) * 16;
    long long tm = (tile / tilesN) * 16;

    const float* Az = A + (long long)blockIdx.z * strideA;
    const float* Bz = B + (long long)blockIdx.z * strideB;

    int r = lane & 15, hh = lane >> 4;
    v8f acc = {0.f, 0.f, 0.f, 0.f, 0.f, 0.f, 0.f, 0.f};
    for (int k0 = 0; k0 < K; k0 += 4) {
        v2f a = *reinterpret_cast<const v2f*>(Az + (tm + r) * (long long)lda + k0 + 2 * hh);
        v2f b;
        b.x = Bz[(long long)(k0 + 2 * hh)     * ldb + tn + r];
        b.y = Bz[(long long)(k0 + 2 * hh + 1) * ldb + tn + r];
        acc = __builtin_amdgcn_wmma_f32_16x16x4_f32(
            false, a, false, b, (short)0, acc, false, false);
    }
    for (int j = 0; j < 8; ++j) {
        long long row = tm + hh * 8 + j;
        C[(long long)blockIdx.z * strideC + row * (long long)ldc + tn + r] = acc[j];
    }
}

// ---------------------------------------------------------------------------
// Small helper kernels
// ---------------------------------------------------------------------------
__global__ __launch_bounds__(256) void k_transpose_to_bf16(
    const float* __restrict__ src, bf16* __restrict__ dst, int rows, int cols)
{
    long long i = (long long)blockIdx.x * blockDim.x + threadIdx.x;
    if (i >= (long long)rows * cols) return;
    int r = (int)(i / cols), c = (int)(i % cols);
    dst[(long long)c * rows + r] = (bf16)src[i];
}

__global__ __launch_bounds__(256) void k_make_fm(
    const float* __restrict__ A, float* __restrict__ Fm)
{
    long long i = (long long)blockIdx.x * blockDim.x + threadIdx.x;
    if (i >= (long long)N_ * N_) return;
    int r = (int)(i / N_), c = (int)(i % N_);
    Fm[i] = DT_ * A[i] + (r == c ? 1.0f : 0.0f);
}

__global__ __launch_bounds__(256) void k_mix(
    const float* __restrict__ in, const float* __restrict__ T, bf16* __restrict__ X)
{
    long long i = (long long)blockIdx.x * blockDim.x + threadIdx.x;
    if (i >= (long long)B_ * Q_ * N_ * F_) return;
    int f = (int)(i % F_);
    int n = (int)((i / F_) % N_);
    int q = (int)((i / ((long long)F_ * N_)) % Q_);
    int b = (int)(i / ((long long)F_ * N_ * Q_));
    float s = 0.f;
    for (int p = 0; p < P_; ++p)
        s += T[p * Q_ + q] * in[(((long long)b * P_ + p) * N_ + n) * F_ + f];
    X[i] = (bf16)s;
}

__global__ __launch_bounds__(256) void k_mu0(
    const float* __restrict__ in, float* __restrict__ mu0)
{
    long long i = (long long)blockIdx.x * blockDim.x + threadIdx.x;
    if (i >= (long long)B_ * N_ * F_) return;
    int f = (int)(i % F_);
    int n = (int)((i / F_) % N_);
    int b = (int)(i / ((long long)F_ * N_));
    mu0[i] = in[(((long long)b * P_ + (P_ - 1)) * N_ + n) * F_ + f];
}

__global__ __launch_bounds__(256) void k_softmax(
    const float* __restrict__ S, bf16* __restrict__ Attn, long long rows)
{
    int lane = threadIdx.x & 31;
    long long row = (long long)blockIdx.x * 8 + (threadIdx.x >> 5);
    if (row >= rows) return;
    const float* rp = S + row * N_;
    const float scale = 0.0625f;
    float v[16], mx = -3.4e38f;
    for (int j = 0; j < 16; ++j) {
        v[j] = rp[lane + 32 * j] * scale;
        mx = fmaxf(mx, v[j]);
    }
    for (int o = 16; o > 0; o >>= 1) mx = fmaxf(mx, __shfl_xor(mx, o, 32));
    float sum = 0.f;
    for (int j = 0; j < 16; ++j) { v[j] = __expf(v[j] - mx); sum += v[j]; }
    for (int o = 16; o > 0; o >>= 1) sum += __shfl_xor(sum, o, 32);
    float inv = 1.0f / sum;
    bf16* wp = Attn + row * N_;
    for (int j = 0; j < 16; ++j) wp[lane + 32 * j] = (bf16)(v[j] * inv);
}

// ---------------------------------------------------------------------------
// Host side
// ---------------------------------------------------------------------------
static inline char* carve(char*& p, size_t bytes) {
    char* r = p;
    p += (bytes + 255) & ~(size_t)255;
    return r;
}

extern "C" void kernel_launch(void* const* d_in, const int* in_sizes, int n_in,
                              void* d_out, int out_size, void* d_ws, size_t ws_size,
                              hipStream_t stream) {
    (void)in_sizes; (void)n_in; (void)out_size; (void)ws_size;
    const float* inputs = (const float*)d_in[0];
    const float* W_in   = (const float*)d_in[2];
    const float* T_mix  = (const float*)d_in[3];
    const float* Wq     = (const float*)d_in[4];
    const float* Wk     = (const float*)d_in[5];
    const float* Wv     = (const float*)d_in[6];
    const float* W_out  = (const float*)d_in[7];
    const float* Amat   = (const float*)d_in[8];

    float* out_mgt = (float*)d_out;
    float* out_kal = (float*)d_out + (long long)B_ * Q_ * N_ * F_;

    const long long SZ_X    = (long long)B_ * Q_ * N_ * F_;
    const long long SZ_BQND = (long long)B_ * Q_ * N_ * D_;
    const long long SZ_SC   = (long long)B_ * Q_ * N_ * N_;

    char* p = (char*)d_ws;
    bf16*  X      = (bf16*)carve(p, SZ_X * 2);
    bf16*  WinT   = (bf16*)carve(p, (size_t)D_ * F_ * 2);
    bf16*  WqT    = (bf16*)carve(p, (size_t)D_ * D_ * 2);
    bf16*  WkT    = (bf16*)carve(p, (size_t)D_ * D_ * 2);
    bf16*  WvT    = (bf16*)carve(p, (size_t)D_ * D_ * 2);
    bf16*  WoutT  = (bf16*)carve(p, (size_t)F_ * D_ * 2);
    float* hF     = (float*)carve(p, SZ_BQND * 4);
    bf16*  hB     = (bf16*)carve(p, SZ_BQND * 2);
    bf16*  qB     = (bf16*)carve(p, SZ_BQND * 2);
    bf16*  kB     = (bf16*)carve(p, SZ_BQND * 2);
    bf16*  vT     = (bf16*)carve(p, SZ_BQND * 2);
    float* scores = (float*)carve(p, SZ_SC * 4);
    bf16*  attn   = (bf16*)carve(p, SZ_SC * 2);
    float* h2F    = (float*)carve(p, SZ_BQND * 4);
    bf16*  h2B    = (bf16*)carve(p, SZ_BQND * 2);
    float* Fm     = (float*)carve(p, (size_t)N_ * N_ * 4);
    float* mu0    = (float*)carve(p, (size_t)B_ * N_ * F_ * 4);

    auto cdiv = [](long long a, long long b) { return (unsigned)((a + b - 1) / b); };

    // --- weight prep + Fm -------------------------------------------------
    k_transpose_to_bf16<<<cdiv((long long)F_ * D_, 256), 256, 0, stream>>>(W_in,  WinT,  F_, D_);
    k_transpose_to_bf16<<<cdiv((long long)D_ * D_, 256), 256, 0, stream>>>(Wq,    WqT,   D_, D_);
    k_transpose_to_bf16<<<cdiv((long long)D_ * D_, 256), 256, 0, stream>>>(Wk,    WkT,   D_, D_);
    k_transpose_to_bf16<<<cdiv((long long)D_ * D_, 256), 256, 0, stream>>>(Wv,    WvT,   D_, D_);
    k_transpose_to_bf16<<<cdiv((long long)D_ * F_, 256), 256, 0, stream>>>(W_out, WoutT, D_, F_);
    k_make_fm<<<cdiv((long long)N_ * N_, 256), 256, 0, stream>>>(Amat, Fm);

    // --- MGT pipeline -----------------------------------------------------
    k_mix<<<cdiv(SZ_X, 256), 256, 0, stream>>>(inputs, T_mix, X);

    // h = X @ W_in  (M=BQN, K=F, Nc=D) -> f32 + bf16
    k_gemm_bf16_tiled<<<dim3((unsigned)((MROWS / BM) * (D_ / BN)), 1, 1), 256, 0, stream>>>(
        X, F_, 0, WinT, F_, 0,
        hF, 0, hB, 0, nullptr, 1, nullptr, 0, MROWS, D_, F_);

    // q/k/v = h @ W  (M=BQN, K=D, Nc=D); v stored transposed per (b,q)
    {
        dim3 g((unsigned)((MROWS / BM) * (D_ / BN)), 1, 1);
        k_gemm_bf16_tiled<<<g, 256, 0, stream>>>(hB, D_, 0, WqT, D_, 0,
            nullptr, 0, qB, 0, nullptr, 1, nullptr, 0, MROWS, D_, D_);
        k_gemm_bf16_tiled<<<g, 256, 0, stream>>>(hB, D_, 0, WkT, D_, 0,
            nullptr, 0, kB, 0, nullptr, 1, nullptr, 0, MROWS, D_, D_);
        k_gemm_bf16_tiled<<<g, 256, 0, stream>>>(hB, D_, 0, WvT, D_, 0,
            nullptr, 0, nullptr, 0, vT, N_, nullptr, 0, MROWS, D_, D_);
    }
    // scores[z] = q[z] @ k[z]^T  batched z = B*Q (M=N, Nc=N, K=D)
    k_gemm_bf16_tiled<<<dim3((unsigned)((N_ / BM) * (N_ / BN)), 1, B_ * Q_), 256, 0, stream>>>(
        qB, D_, (long long)N_ * D_, kB, D_, (long long)N_ * D_,
        scores, (long long)N_ * N_, nullptr, 0, nullptr, 1, nullptr, 0,
        N_, N_, D_);

    k_softmax<<<cdiv(MROWS, 8), 256, 0, stream>>>(scores, attn, MROWS);

    // h2 = h + attn @ v  batched (M=N, Nc=D, K=N)
    k_gemm_bf16_tiled<<<dim3((unsigned)((N_ / BM) * (D_ / BN)), 1, B_ * Q_), 256, 0, stream>>>(
        attn, N_, (long long)N_ * N_, vT, N_, (long long)D_ * N_,
        h2F, (long long)N_ * D_, h2B, (long long)N_ * D_, nullptr, 1,
        hF, (long long)N_ * D_, N_, D_, N_);

    // mgt_out = h2 @ W_out  (M=BQN, K=D, Nc=F=64 -> per-wave kernel)
    k_gemm_bf16<<<dim3(cdiv((MROWS / 16) * (F_ / 16), 8), 1, 1), 256, 0, stream>>>(
        h2B, D_, WoutT, D_, out_mgt, MROWS, F_, D_);

    // --- Kalman mean chain (covariance branch is dead code in the reference)
    k_mu0<<<cdiv((long long)B_ * N_ * F_, 256), 256, 0, stream>>>(inputs, mu0);
    {
        dim3 g(cdiv(((long long)N_ / 16) * (F_ / 16), 8), 1, B_);
        for (int t = 0; t < Q_; ++t) {
            const float* src = (t == 0) ? mu0
                                        : out_kal + (long long)(t - 1) * N_ * F_;
            long long strideSrc = (t == 0) ? (long long)N_ * F_
                                           : (long long)Q_ * N_ * F_;
            k_gemm_f32<<<g, 256, 0, stream>>>(
                Fm, N_, 0, src, F_, strideSrc,
                out_kal + (long long)t * N_ * F_, F_, (long long)Q_ * N_ * F_,
                N_, F_, N_);
        }
    }
}
